// Attention_59923383713843
// MI455X (gfx1250) — matrix operand order repaired and checked by reading, not compiled, not run on hardware
//
#include <hip/hip_runtime.h>
#include <hip/hip_bf16.h>
#include <math.h>

#define DIM   768
#define NHEAD 12
#define HDIM  64
#define SEQ   2048
#define BATCH 2
#define MTOT  (BATCH * SEQ)   // 4096 rows in the token dimension

typedef __attribute__((ext_vector_type(16))) __bf16 v16bf;
typedef __attribute__((ext_vector_type(8)))  __bf16 v8bf;
typedef __attribute__((ext_vector_type(8)))  float  v8f;
typedef int v4i __attribute__((vector_size(16)));   // matches builtin's expected pointee

union F8 { v8f v; float f[8]; };

#if defined(__gfx1250__) && __has_builtin(__builtin_amdgcn_global_load_async_to_lds_b128)
#define HAVE_ASYNC_LDS 1
typedef __attribute__((address_space(1))) v4i* gptr_v4i;
typedef __attribute__((address_space(3))) v4i* lptr_v4i;
#else
#define HAVE_ASYNC_LDS 0
#endif

static __device__ __forceinline__ unsigned short f32_to_bf16(float x) {
  unsigned int u = __builtin_bit_cast(unsigned int, x);
  u += 0x7FFFu + ((u >> 16) & 1u);   // round-to-nearest-even
  return (unsigned short)(u >> 16);
}

static __device__ __forceinline__ v8f zero_v8f() {
  v8f z = {0.f, 0.f, 0.f, 0.f, 0.f, 0.f, 0.f, 0.f};
  return z;
}

// A fragment: 16(M) x 32(K) bf16, source row-major with leading dim `ld` (K contiguous).
// ISA layout: lane<16 -> M=lane, halfs 0..7 = K0..7, halfs 8..15 = K16..23; lane>=16 -> K+8.
static __device__ __forceinline__ v16bf load_a_frag(const unsigned short* base, int ld, int lane) {
  const int half = lane >> 4;
  const int r    = lane & 15;
  const unsigned short* row = base + (size_t)r * ld + half * 8;
  union { v16bf v; v8bf h[2]; } u;
  u.h[0] = *(const v8bf*)(row);        // K = half*8 + 0..7
  u.h[1] = *(const v8bf*)(row + 16);   // K = 16 + half*8 + 0..7
  return u.v;
}

// B fragment: 32(K) x 16(N) bf16, source stored as [N][K] row-major with leading dim `ld`.
// ISA layout: lane<16 -> N=lane, halfs 0..7 = K0..7, halfs 8..15 = K16..23; lane>=16 -> N=lane-16, K+8.
static __device__ __forceinline__ v16bf load_b_frag(const unsigned short* base, int ld, int lane) {
  const int half = lane >> 4;
  const int c    = lane & 15;
  const unsigned short* row = base + (size_t)c * ld + half * 8;
  union { v16bf v; v8bf h[2]; } u;
  u.h[0] = *(const v8bf*)(row);
  u.h[1] = *(const v8bf*)(row + 16);
  return u.v;
}

static __device__ __forceinline__ v8f wmma_bf16(v16bf a, v16bf b, v8f c) {
  return __builtin_amdgcn_wmma_f32_16x16x32_bf16(false, a, false, b, (short)0, c, false, false);
}

// 16-byte global->LDS copy: async (ASYNCcnt, no VGPR round-trip) when available.
static __device__ __forceinline__ void copy16_g2l(const unsigned short* g, unsigned short* l) {
#if HAVE_ASYNC_LDS
  __builtin_amdgcn_global_load_async_to_lds_b128(
      (gptr_v4i)(void*)(unsigned short*)g,
      (lptr_v4i)(void*)l, 0, 0);
#else
  *(v8bf*)l = *(const v8bf*)g;
#endif
}

static __device__ __forceinline__ void wait_async_copies() {
#if HAVE_ASYNC_LDS
#if __has_builtin(__builtin_amdgcn_s_wait_asynccnt)
  __builtin_amdgcn_s_wait_asynccnt(0);
#else
  asm volatile("s_wait_asynccnt 0x0" ::: "memory");
#endif
#endif
}

// ---------------------------------------------------------------- fp32 -> bf16
__global__ void cvt_bf16(const float* __restrict__ in, unsigned short* __restrict__ out, int n) {
  int i = blockIdx.x * blockDim.x + threadIdx.x;
  int stride = gridDim.x * blockDim.x;
  for (; i < n; i += stride) out[i] = f32_to_bf16(in[i]);
}

// ------------------------------------------------- QKV GEMM: [4096x768]x[2304x768]^T
// 32x32 tile per wave (2x2 WMMA accumulators) for 1 fragment-load per WMMA.
// Writes Q (pre-scaled by 1/sqrt(Dh)) and K as [B,H,N,Dh], V transposed as [B,H,Dh,N].
__global__ void qkv_gemm(const unsigned short* __restrict__ xb,
                         const unsigned short* __restrict__ wb,
                         unsigned short* __restrict__ Q,
                         unsigned short* __restrict__ Kk,
                         unsigned short* __restrict__ Vt) {
  const int lane   = threadIdx.x & 31;
  const int wave   = threadIdx.x >> 5;
  const int tile_m = blockIdx.x * 32;
  const int tile_o = (blockIdx.y * 4 + wave) * 32;

  F8 acc[2][2];
#pragma unroll
  for (int i = 0; i < 2; ++i)
#pragma unroll
    for (int j = 0; j < 2; ++j) acc[i][j].v = zero_v8f();

  const unsigned short* a0p = xb + (size_t)tile_m * DIM;
  const unsigned short* a1p = a0p + (size_t)16 * DIM;
  const unsigned short* b0p = wb + (size_t)tile_o * DIM;
  const unsigned short* b1p = b0p + (size_t)16 * DIM;
  for (int k0 = 0; k0 < DIM; k0 += 32) {
    const v16bf A0 = load_a_frag(a0p + k0, DIM, lane);
    const v16bf A1 = load_a_frag(a1p + k0, DIM, lane);
    const v16bf B0 = load_b_frag(b0p + k0, DIM, lane);
    const v16bf B1 = load_b_frag(b1p + k0, DIM, lane);
    acc[0][0].v = wmma_bf16(A0, B0, acc[0][0].v);
    acc[0][1].v = wmma_bf16(A0, B1, acc[0][1].v);
    acc[1][0].v = wmma_bf16(A1, B0, acc[1][0].v);
    acc[1][1].v = wmma_bf16(A1, B1, acc[1][1].v);
  }

  const int half = lane >> 4;
  const int c    = lane & 15;
#pragma unroll
  for (int oj = 0; oj < 2; ++oj) {
    const int col   = tile_o + oj * 16 + c;
    const int third = col / DIM;      // 0=Q, 1=K, 2=V
    const int o     = col % DIM;
    const int h     = o / HDIM;
    const int d     = o % HDIM;
#pragma unroll
    for (int mi = 0; mi < 2; ++mi) {
#pragma unroll
      for (int r = 0; r < 8; ++r) {
        const int mrow = tile_m + mi * 16 + r + 8 * half;
        const int b_   = mrow >> 11;           // /SEQ
        const int n    = mrow & (SEQ - 1);
        const int bh   = b_ * NHEAD + h;
        const float v  = acc[mi][oj].f[r];
        if (third == 0)
          Q[((size_t)bh * SEQ + n) * HDIM + d] = f32_to_bf16(v * 0.125f);  // Dh^-0.5
        else if (third == 1)
          Kk[((size_t)bh * SEQ + n) * HDIM + d] = f32_to_bf16(v);
        else
          Vt[((size_t)bh * HDIM + d) * SEQ + n] = f32_to_bf16(v);
      }
    }
  }
}

// ----------------------------------------------- fused flash attention per (b,h)
// Block = 8 waves, all on the same (b,h): K/V tiles staged once per m-step into
// LDS (async global->LDS when available), each wave owns a 16-query tile.
__global__ void flash_attn(const unsigned short* __restrict__ Q,
                           const unsigned short* __restrict__ Kk,
                           const unsigned short* __restrict__ Vt,
                           const int* __restrict__ mask,
                           unsigned short* __restrict__ O) {
  __shared__ __align__(16) unsigned short kT[32 * HDIM];      // [m_local][d]   4KB
  __shared__ __align__(16) unsigned short vT[HDIM * 32];      // [d][m_local]   4KB
  __shared__ __align__(16) unsigned short ldsP[8][16 * 32];   // P tile / wave  8KB

  const int tid  = threadIdx.x;
  const int lane = tid & 31;
  const int wave = tid >> 5;
  const int bh   = blockIdx.x;
  const int b    = bh / NHEAD;
  const int h    = bh % NHEAD;
  const int n0   = (blockIdx.y * 8 + wave) * 16;

  const size_t base = (size_t)bh * SEQ * HDIM;
  const unsigned short* Qb = Q  + base + (size_t)n0 * HDIM;
  const unsigned short* Kb = Kk + base;            // [SEQ][HDIM]
  const unsigned short* Vb = Vt + base;            // [HDIM][SEQ]
  const int half = lane >> 4;
  const int c    = lane & 15;

  // cooperative-copy chunk coordinates (16B per thread per tile)
  const int krow = tid >> 3, kcol = (tid & 7) * 8;   // K tile: 32 rows x 128B
  const int vrow = tid >> 2, vcol = (tid & 3) * 8;   // V tile: 64 rows x 64B

  const v16bf qa0 = load_a_frag(Qb, HDIM, lane);        // d = 0..31
  const v16bf qa1 = load_a_frag(Qb + 32, HDIM, lane);   // d = 32..63

  F8 oacc[4];
#pragma unroll
  for (int j = 0; j < 4; ++j) oacc[j].v = zero_v8f();
  float mrun[8], lrun[8];
#pragma unroll
  for (int r = 0; r < 8; ++r) { mrun[r] = -INFINITY; lrun[r] = 0.f; }

  unsigned short* pw = &ldsP[wave][0];

  for (int m0 = 0; m0 < SEQ; m0 += 32) {
    __syncthreads();   // previous iteration's readers done before overwrite
    copy16_g2l(Kb + (size_t)(m0 + krow) * HDIM + kcol, &kT[krow * HDIM + kcol]);
    copy16_g2l(Vb + (size_t)vrow * SEQ + m0 + vcol,    &vT[vrow * 32 + vcol]);
    wait_async_copies();
    __syncthreads();

    // S tile (16 x 32) = Q(16x64) @ K^T, fragments straight from LDS
    F8 slo, shi;
    slo.v = wmma_bf16(qa0, load_b_frag(kT,                 HDIM, lane), zero_v8f());
    slo.v = wmma_bf16(qa1, load_b_frag(kT + 32,            HDIM, lane), slo.v);
    shi.v = wmma_bf16(qa0, load_b_frag(kT + 16 * HDIM,      HDIM, lane), zero_v8f());
    shi.v = wmma_bf16(qa1, load_b_frag(kT + 16 * HDIM + 32, HDIM, lane), shi.v);

    // key-side mask (columns)
    const int mlo = mask[b * SEQ + m0 + c];
    const int mhi = mask[b * SEQ + m0 + 16 + c];
#pragma unroll
    for (int r = 0; r < 8; ++r) {
      if (mlo == 0) slo.f[r] = -INFINITY;
      if (mhi == 0) shi.f[r] = -INFINITY;
    }

    // online softmax: rows live as 8 regs/lane, reduce over the 16 lanes of each half
    float alpha[8];
    F8 plo, phi;
#pragma unroll
    for (int r = 0; r < 8; ++r) {
      float t = fmaxf(slo.f[r], shi.f[r]);
      t = fmaxf(t, __shfl_xor(t, 1, 32));
      t = fmaxf(t, __shfl_xor(t, 2, 32));
      t = fmaxf(t, __shfl_xor(t, 4, 32));
      t = fmaxf(t, __shfl_xor(t, 8, 32));
      const float mnew = fmaxf(mrun[r], t);
      alpha[r] = __expf(mrun[r] - mnew);
      mrun[r]  = mnew;
      const float pl = __expf(slo.f[r] - mnew);
      const float ph = __expf(shi.f[r] - mnew);
      plo.f[r] = pl; phi.f[r] = ph;
      float rs = pl + ph;
      rs += __shfl_xor(rs, 1, 32);
      rs += __shfl_xor(rs, 2, 32);
      rs += __shfl_xor(rs, 4, 32);
      rs += __shfl_xor(rs, 8, 32);
      lrun[r] = lrun[r] * alpha[r] + rs;
    }

    // transpose P from C-layout to A-layout through per-wave LDS (bf16)
#pragma unroll
    for (int r = 0; r < 8; ++r) {
      const int row = r + 8 * half;
      pw[row * 32 + c]      = f32_to_bf16(plo.f[r]);
      pw[row * 32 + 16 + c] = f32_to_bf16(phi.f[r]);
    }
    const v16bf pa = load_a_frag(pw, 32, lane);

    // O = diag(alpha)*O + P @ V   (V fragments from LDS, K-contiguous)
#pragma unroll
    for (int j = 0; j < 4; ++j) {
#pragma unroll
      for (int r = 0; r < 8; ++r) oacc[j].f[r] *= alpha[r];
      const v16bf vb = load_b_frag(vT + j * 16 * 32, 32, lane);
      oacc[j].v = wmma_bf16(pa, vb, oacc[j].v);
    }
  }

  // finalize and store as bf16 [B, N, H*Dh]
#pragma unroll
  for (int j = 0; j < 4; ++j)
#pragma unroll
    for (int r = 0; r < 8; ++r) {
      const int n = n0 + r + 8 * half;
      const float val = oacc[j].f[r] / lrun[r];
      O[((size_t)(b * SEQ + n)) * DIM + h * HDIM + j * 16 + c] = f32_to_bf16(val);
    }
}

// ------------------------------------------------- proj GEMM + bias -> fp32 out
__global__ void proj_gemm(const unsigned short* __restrict__ ab,
                          const unsigned short* __restrict__ wb,
                          const float* __restrict__ bias,
                          float* __restrict__ out) {
  const int lane   = threadIdx.x & 31;
  const int wave   = threadIdx.x >> 5;
  const int tile_m = blockIdx.x * 32;
  const int tile_o = (blockIdx.y * 4 + wave) * 32;

  F8 acc[2][2];
#pragma unroll
  for (int i = 0; i < 2; ++i)
#pragma unroll
    for (int j = 0; j < 2; ++j) acc[i][j].v = zero_v8f();

  const unsigned short* a0p = ab + (size_t)tile_m * DIM;
  const unsigned short* a1p = a0p + (size_t)16 * DIM;
  const unsigned short* b0p = wb + (size_t)tile_o * DIM;
  const unsigned short* b1p = b0p + (size_t)16 * DIM;
  for (int k0 = 0; k0 < DIM; k0 += 32) {
    const v16bf A0 = load_a_frag(a0p + k0, DIM, lane);
    const v16bf A1 = load_a_frag(a1p + k0, DIM, lane);
    const v16bf B0 = load_b_frag(b0p + k0, DIM, lane);
    const v16bf B1 = load_b_frag(b1p + k0, DIM, lane);
    acc[0][0].v = wmma_bf16(A0, B0, acc[0][0].v);
    acc[0][1].v = wmma_bf16(A0, B1, acc[0][1].v);
    acc[1][0].v = wmma_bf16(A1, B0, acc[1][0].v);
    acc[1][1].v = wmma_bf16(A1, B1, acc[1][1].v);
  }

  const int half = lane >> 4;
  const int c    = lane & 15;
#pragma unroll
  for (int oj = 0; oj < 2; ++oj) {
    const int col = tile_o + oj * 16 + c;
    const float bv = bias[col];
#pragma unroll
    for (int mi = 0; mi < 2; ++mi) {
#pragma unroll
      for (int r = 0; r < 8; ++r) {
        const int mrow = tile_m + mi * 16 + r + 8 * half;
        out[(size_t)mrow * DIM + col] = acc[mi][oj].f[r] + bv;
      }
    }
  }
}

extern "C" void kernel_launch(void* const* d_in, const int* in_sizes, int n_in,
                              void* d_out, int out_size, void* d_ws, size_t ws_size,
                              hipStream_t stream) {
  (void)in_sizes; (void)n_in; (void)out_size; (void)ws_size;
  const float* x        = (const float*)d_in[0];
  const int*   att_mask = (const int*)d_in[1];
  const float* qkv_w    = (const float*)d_in[2];
  const float* proj_w   = (const float*)d_in[3];
  const float* proj_b   = (const float*)d_in[4];
  float*       out      = (float*)d_out;

  // workspace layout (bf16 halves), total ~36.2 MB
  unsigned short* ws    = (unsigned short*)d_ws;
  unsigned short* xb    = ws;                                  // 4096*768
  unsigned short* wqkv  = xb    + (size_t)MTOT * DIM;          // 2304*768
  unsigned short* wproj = wqkv  + (size_t)3 * DIM * DIM;       // 768*768
  unsigned short* Qb    = wproj + (size_t)DIM * DIM;           // [B,H,N,Dh]
  unsigned short* Kb    = Qb    + (size_t)MTOT * DIM;          // [B,H,N,Dh]
  unsigned short* Vt    = Kb    + (size_t)MTOT * DIM;          // [B,H,Dh,N]
  unsigned short* ao    = Vt    + (size_t)MTOT * DIM;          // [B,N,H*Dh]

  cvt_bf16<<<2048, 256, 0, stream>>>(x,      xb,    MTOT * DIM);
  cvt_bf16<<<2048, 256, 0, stream>>>(qkv_w,  wqkv,  3 * DIM * DIM);
  cvt_bf16<<<1024, 256, 0, stream>>>(proj_w, wproj, DIM * DIM);

  qkv_gemm <<<dim3(MTOT / 32, (3 * DIM) / 128), 128, 0, stream>>>(xb, wqkv, Qb, Kb, Vt);
  flash_attn<<<dim3(BATCH * NHEAD, SEQ / 128), 256, 0, stream>>>(Qb, Kb, Vt, att_mask, ao);
  proj_gemm<<<dim3(MTOT / 32, DIM / 128), 128, 0, stream>>>(ao, wproj, proj_b, out);
}
